// DopplerAttention_19301583028928
// MI455X (gfx1250) — matrix operand order repaired and checked by reading, not compiled, not run on hardware
//
#include <hip/hip_runtime.h>
#include <hip/hip_bf16.h>

// ---------------------------------------------------------------------------
// DopplerAttention for MI455X (gfx1250, wave32, WMMA).
//
// Pipeline (all matmul stages via v_wmma_f32_16x16x32_bf16, fp32 accum):
//   K1: qkv(bf16)[b][768][16384] = qkv_w @ x + qkv_b          (GEMM 768x16384x256, x8)
//   K2: per (b,hd,c): 4 attention problems (wb=0..3), S=QtK (128x128x32),
//       fp32 softmax, O=P@Vt (128x32x128); packed bf16 stores to y[b][256][H][W]
//   K3: out(fp32) = proj_w @ y + proj_b                        (GEMM 256x16384x256, x8)
//
// Workspace: qkv bf16 = 201.3 MB, y bf16 = 67.1 MB  (268 MB total).
// All transposing LDS writes use packed 2xbf16 b32 stores (v_cvt_pk_bf16_f32).
// ---------------------------------------------------------------------------

typedef __attribute__((ext_vector_type(16))) __bf16 v16bf;
typedef __attribute__((ext_vector_type(8)))  __bf16 v8bf;
typedef __attribute__((ext_vector_type(2)))  __bf16 v2bf;
typedef __attribute__((ext_vector_type(8)))  float  v8f;

#define HW    16384   // 128*128
#define HDIM  128
#define WDIM  128

static __device__ inline v16bf mk16(v8bf lo, v8bf hi) {
    v16bf r;
#pragma unroll
    for (int i = 0; i < 8; ++i) { r[i] = lo[i]; r[i + 8] = hi[i]; }
    return r;
}
static __device__ inline v2bf pk2(float a, float b) {
    v2bf r; r[0] = (__bf16)a; r[1] = (__bf16)b; return r;   // v_cvt_pk_bf16_f32
}

// A fragment (16x32 bf16, ISA 7.12.2): lane L holds row M=L%16;
// lane-half selects K-blocks {kb..kb+7, kb+16..kb+23}, kb = (L>>4)*8.
static __device__ inline v16bf fragA(const __bf16* rowPtr, int lane) {
    const __bf16* p = rowPtr + ((lane >> 4) * 8);
    return mk16(*(const v8bf*)p, *(const v8bf*)(p + 16));
}
// B fragment (32x16 bf16): lane L holds col N=L%16; lane-half selects
// K-blocks {kb..kb+7, kb+16..kb+23}, kb = (L>>4)*8 (same as A).
static __device__ inline v16bf fragB(const __bf16* colPtr, int lane) {
    const __bf16* p = colPtr + ((lane >> 4) * 8);
    return mk16(*(const v8bf*)p, *(const v8bf*)(p + 16));
}

// ---------------------------------------------------------------------------
// Kernel 1: qkv GEMM.  out[b][o][hw] = sum_c W[o][c] * x[b][c][hw] + bias[o]
// Block: 256 threads (8 waves). Output tile 128(M=o) x 128(N=hw), K step 32.
// Waves: 4(M groups of 32) x 2(N groups of 64); each wave owns 2x4 fragments.
// ---------------------------------------------------------------------------
#define LDAB 40   // padded LDS row stride (bf16 elems); 80B, 16B-aligned

__global__ void qkv_gemm_kernel(const float* __restrict__ X,
                                const float* __restrict__ Wq,
                                const float* __restrict__ bias,
                                __bf16* __restrict__ out) {
    extern __shared__ char smem[];
    __bf16* At = (__bf16*)smem;            // [128][LDAB]
    __bf16* Bt = At + 128 * LDAB;          // [128][LDAB]
    float*  biasS = (float*)(Bt + 128 * LDAB);  // [128]

    const int tid = threadIdx.x, lane = tid & 31, wave = tid >> 5;
    const int wm = wave & 3, wn = wave >> 2;
    const int nt = blockIdx.x * 128, mt = blockIdx.y * 128, b = blockIdx.z;
    const float* Xb = X + (size_t)b * 256 * HW;

    if (tid < 128) biasS[tid] = bias[mt + tid];

    v8f acc[2][4];
#pragma unroll
    for (int i = 0; i < 2; ++i)
#pragma unroll
        for (int j = 0; j < 4; ++j) acc[i][j] = (v8f){};

    for (int k0 = 0; k0 < 256; k0 += 32) {
        __syncthreads();
        {   // A tile: Wq[mt+m][k0+k], fp32 -> bf16, row-major [m][k]
            int m = tid >> 1, kh = (tid & 1) * 16;
            const float* s = Wq + (size_t)(mt + m) * 256 + k0 + kh;
            __bf16* d = At + m * LDAB + kh;
#pragma unroll
            for (int i = 0; i < 16; i += 4) {
                float4 f = *(const float4*)(s + i);
                d[i] = (__bf16)f.x; d[i + 1] = (__bf16)f.y;
                d[i + 2] = (__bf16)f.z; d[i + 3] = (__bf16)f.w;
            }
        }
        {   // B tile (transposed): X[k0+{2kp,2kp+1}][nt+n] -> Bt[n][2kp..2kp+1]
            // packed pairs along k -> 8x ds_store_b32 per thread
            int kp = tid & 15, n0 = (tid >> 4) * 8;
            const float* s0 = Xb + (size_t)(k0 + 2 * kp) * HW + nt + n0;
            const float* s1 = s0 + HW;
            if (k0 + 32 < 256)  // CDNA5 global_prefetch_b8 on next K slice
                __builtin_prefetch(s0 + (size_t)32 * HW, 0, 1);
            float4 a0 = *(const float4*)(s0),     a1 = *(const float4*)(s0 + 4);
            float4 b0 = *(const float4*)(s1),     b1 = *(const float4*)(s1 + 4);
            __bf16* d = Bt + 2 * kp;
            *(v2bf*)(d + (n0    ) * LDAB) = pk2(a0.x, b0.x);
            *(v2bf*)(d + (n0 + 1) * LDAB) = pk2(a0.y, b0.y);
            *(v2bf*)(d + (n0 + 2) * LDAB) = pk2(a0.z, b0.z);
            *(v2bf*)(d + (n0 + 3) * LDAB) = pk2(a0.w, b0.w);
            *(v2bf*)(d + (n0 + 4) * LDAB) = pk2(a1.x, b1.x);
            *(v2bf*)(d + (n0 + 5) * LDAB) = pk2(a1.y, b1.y);
            *(v2bf*)(d + (n0 + 6) * LDAB) = pk2(a1.z, b1.z);
            *(v2bf*)(d + (n0 + 7) * LDAB) = pk2(a1.w, b1.w);
        }
        __syncthreads();

        v16bf afr[2];
#pragma unroll
        for (int i = 0; i < 2; ++i)
            afr[i] = fragA(At + (wm * 32 + i * 16 + (lane & 15)) * LDAB, lane);
#pragma unroll
        for (int j = 0; j < 4; ++j) {
            v16bf bfr = fragB(Bt + (wn * 64 + j * 16 + (lane & 15)) * LDAB, lane);
#pragma unroll
            for (int i = 0; i < 2; ++i)
                acc[i][j] = __builtin_amdgcn_wmma_f32_16x16x32_bf16(
                    false, afr[i], false, bfr, (short)0, acc[i][j], false, false);
        }
    }

    // epilogue: + bias, bf16 store (C/D layout: lane holds col N=lane%16,
    // rows M = (lane>>4)*8 + r for vector element r)
    __bf16* ob = out + (size_t)b * 768 * HW;
#pragma unroll
    for (int i = 0; i < 2; ++i) {
        int ml = wm * 32 + i * 16 + (lane >> 4) * 8;
#pragma unroll
        for (int j = 0; j < 4; ++j) {
            int n = nt + wn * 64 + j * 16 + (lane & 15);
#pragma unroll
            for (int r = 0; r < 8; ++r) {
                float v = acc[i][j][r] + biasS[ml + r];
                ob[(size_t)(mt + ml + r) * HW + n] = (__bf16)v;
            }
        }
    }
}

// ---------------------------------------------------------------------------
// Kernel 2: attention. One block per (b, hd, c); loops wb=0..3.
//   Q/K tiles [hh][c'] 128x32, V transposed [c'][hh] 32x128  (bf16, LDS)
//   S = Q Kt  (M=x1, N=x2, K=32)  -> fp32 LDS [128][132] (float4-aligned rows)
//   softmax rows (fp32, wave32 shfl), P bf16 LDS [128][136], 1/rowsum in LDS
//   O = P Vt  (M=x1, N=c', K=128) -> scaled, packed into Oa[c'][x1][wb] bf16
//   final: 8B packed stores y[b][c'*8+hd][x1][c*4+wb]
// ---------------------------------------------------------------------------
#define LDQ 40
#define LDV 136
#define LDSM 132
#define LDP 136
#define SMEM_ATTN (128*LDQ*2*2 + 32*LDV*2 + 128*LDSM*4 + 128*LDP*2 + 32*128*4*2 + 128*4)

__global__ void attn_kernel(const __bf16* __restrict__ qkv,
                            __bf16* __restrict__ y) {
    extern __shared__ char smem[];
    __bf16* Qt = (__bf16*)smem;                 // [128][LDQ]
    __bf16* Kt = Qt + 128 * LDQ;                // [128][LDQ]
    __bf16* Vt = Kt + 128 * LDQ;                // [32][LDV]
    float*  Sm = (float*)(Vt + 32 * LDV);       // [128][LDSM]
    __bf16* Pm = (__bf16*)(Sm + 128 * LDSM);    // [128][LDP]
    __bf16* Oa = Pm + 128 * LDP;                // [32][128][4]
    float*  rs = (float*)(Oa + 32 * 128 * 4);   // [128] reciprocal row sums

    const int tid = threadIdx.x, lane = tid & 31, wave = tid >> 5;
    const int id = blockIdx.x;
    const int b = id >> 8, hd = (id >> 5) & 7, c = id & 31;
    const size_t bbase = (size_t)b * 768 * HW;
    const int qc = hd * 32 + c;
    const float scale = 0.17677669529663687f;   // 1/sqrt(32)

    for (int wb = 0; wb < 4; ++wb) {
        __syncthreads();   // protect previous iteration's LDS reads
        {   // load Q,K row-major [hh][c'], V transposed [c'][hh].
            // thread owns an hh-pair x 8 c' so V-transpose stores pack to b32.
            int hh = (tid & 63) * 2, c0 = (tid >> 6) * 8;
            size_t off = (size_t)hh * WDIM + wb * 32 + c0;
            const __bf16* qp = qkv + bbase + (size_t)qc * HW + off;
            const __bf16* kp = qkv + bbase + (size_t)(256 + qc) * HW + off;
            const __bf16* vp = qkv + bbase + (size_t)(512 + qc) * HW + off;
            v8bf q0 = *(const v8bf*)qp, q1 = *(const v8bf*)(qp + WDIM);
            v8bf k0 = *(const v8bf*)kp, k1 = *(const v8bf*)(kp + WDIM);
            v8bf v0 = *(const v8bf*)vp, v1 = *(const v8bf*)(vp + WDIM);
            *(v8bf*)(Qt + (hh    ) * LDQ + c0) = q0;
            *(v8bf*)(Qt + (hh + 1) * LDQ + c0) = q1;
            *(v8bf*)(Kt + (hh    ) * LDQ + c0) = k0;
            *(v8bf*)(Kt + (hh + 1) * LDQ + c0) = k1;
#pragma unroll
            for (int i = 0; i < 8; ++i) {
                v2bf pr; pr[0] = v0[i]; pr[1] = v1[i];
                *(v2bf*)(Vt + (c0 + i) * LDV + hh) = pr;   // packed b32 store
            }
        }
        __syncthreads();

        {   // S = Q * Kt : wave owns M-tile `wave`, loops 8 N-tiles
            v16bf a = fragA(Qt + (wave * 16 + (lane & 15)) * LDQ, lane);
            int m0 = wave * 16 + (lane >> 4) * 8;
#pragma unroll
            for (int j = 0; j < 8; ++j) {
                v16bf bf = fragB(Kt + (j * 16 + (lane & 15)) * LDQ, lane);
                v8f s = (v8f){};
                s = __builtin_amdgcn_wmma_f32_16x16x32_bf16(
                    false, a, false, bf, (short)0, s, false, false);
                int n = j * 16 + (lane & 15);
#pragma unroll
                for (int r = 0; r < 8; ++r) Sm[(m0 + r) * LDSM + n] = s[r];
            }
        }
        __syncthreads();

        {   // softmax: 2 threads per row, 64 cols each (float4 reads);
            // combine partials via wave32 shfl_xor(1)
            int row = tid >> 1, h = tid & 1;
            const float4* Sr = (const float4*)(Sm + row * LDSM + h * 64);
            float mx = -3.4e38f;
#pragma unroll 4
            for (int i = 0; i < 16; ++i) {
                float4 f = Sr[i];
                mx = fmaxf(mx, fmaxf(fmaxf(f.x, f.y), fmaxf(f.z, f.w)));
            }
            mx = fmaxf(mx, __shfl_xor(mx, 1));
            float ms = mx * scale;
            float sum = 0.f;
            __bf16* Pr = Pm + row * LDP + h * 64;
#pragma unroll 4
            for (int i = 0; i < 16; ++i) {
                float4 f = Sr[i];
                float e0 = __expf(f.x * scale - ms), e1 = __expf(f.y * scale - ms);
                float e2 = __expf(f.z * scale - ms), e3 = __expf(f.w * scale - ms);
                sum += (e0 + e1) + (e2 + e3);
                *(v2bf*)(Pr + 4 * i)     = pk2(e0, e1);   // unnormalized
                *(v2bf*)(Pr + 4 * i + 2) = pk2(e2, e3);   // 1/sum folded later
            }
            sum += __shfl_xor(sum, 1);
            if (h == 0) rs[row] = 1.0f / sum;
        }
        __syncthreads();

        {   // O = P * Vt : M=x1(128), N=c'(32), K=x2(128) in 4 WMMA K-steps
            v8f occ[2] = {(v8f){}, (v8f){}};
#pragma unroll
            for (int ks = 0; ks < 4; ++ks) {
                v16bf a = fragA(Pm + (wave * 16 + (lane & 15)) * LDP + ks * 32, lane);
#pragma unroll
                for (int j = 0; j < 2; ++j) {
                    v16bf bf = fragB(Vt + (j * 16 + (lane & 15)) * LDV + ks * 32, lane);
                    occ[j] = __builtin_amdgcn_wmma_f32_16x16x32_bf16(
                        false, a, false, bf, (short)0, occ[j], false, false);
                }
            }
            int m0 = wave * 16 + (lane >> 4) * 8;
#pragma unroll
            for (int j = 0; j < 2; ++j) {
                int cp = j * 16 + (lane & 15);
#pragma unroll
                for (int r = 0; r < 8; ++r) {
                    int x1 = m0 + r;
                    Oa[(cp * 128 + x1) * 4 + wb] = (__bf16)(occ[j][r] * rs[x1]);
                }
            }
        }
    }
    __syncthreads();

    // packed store: y[b][cp*8+hd][x1][c*4 + (0..3)]  (8 bytes per element group)
    {
        __bf16* yb = y + (size_t)b * 256 * HW;
        for (int i = tid; i < 32 * 128; i += 256) {
            int cp = i >> 7, x1 = i & 127;
            unsigned long long pack = *(const unsigned long long*)(Oa + i * 4);
            *(unsigned long long*)(yb + ((size_t)(cp * 8 + hd) * HDIM + x1) * WDIM + c * 4) = pack;
        }
    }
}

// ---------------------------------------------------------------------------
// Kernel 3: proj GEMM. out[b][o][hw] = sum_c Wp[o][c] * y[b][c][hw] + bias[o]
// Same tiling as kernel 1; B source already bf16, output fp32.
// ---------------------------------------------------------------------------
__global__ void proj_gemm_kernel(const __bf16* __restrict__ Y,
                                 const float* __restrict__ Wp,
                                 const float* __restrict__ bias,
                                 float* __restrict__ out) {
    extern __shared__ char smem[];
    __bf16* At = (__bf16*)smem;
    __bf16* Bt = At + 128 * LDAB;
    float*  biasS = (float*)(Bt + 128 * LDAB);

    const int tid = threadIdx.x, lane = tid & 31, wave = tid >> 5;
    const int wm = wave & 3, wn = wave >> 2;
    const int nt = blockIdx.x * 128, mt = blockIdx.y * 128, b = blockIdx.z;
    const __bf16* Yb = Y + (size_t)b * 256 * HW;

    if (tid < 128) biasS[tid] = bias[mt + tid];

    v8f acc[2][4];
#pragma unroll
    for (int i = 0; i < 2; ++i)
#pragma unroll
        for (int j = 0; j < 4; ++j) acc[i][j] = (v8f){};

    for (int k0 = 0; k0 < 256; k0 += 32) {
        __syncthreads();
        {   // A tile: Wp fp32 -> bf16
            int m = tid >> 1, kh = (tid & 1) * 16;
            const float* s = Wp + (size_t)(mt + m) * 256 + k0 + kh;
            __bf16* d = At + m * LDAB + kh;
#pragma unroll
            for (int i = 0; i < 16; i += 4) {
                float4 f = *(const float4*)(s + i);
                d[i] = (__bf16)f.x; d[i + 1] = (__bf16)f.y;
                d[i + 2] = (__bf16)f.z; d[i + 3] = (__bf16)f.w;
            }
        }
        {   // B tile (transposed): Y[k0+{2kp,2kp+1}][nt+n] -> packed b32 stores
            int kp = tid & 15, n0 = (tid >> 4) * 8;
            const __bf16* s0 = Yb + (size_t)(k0 + 2 * kp) * HW + nt + n0;
            if (k0 + 32 < 256)
                __builtin_prefetch(s0 + (size_t)32 * HW, 0, 1);
            v8bf r0 = *(const v8bf*)s0, r1 = *(const v8bf*)(s0 + HW);
            __bf16* d = Bt + 2 * kp;
#pragma unroll
            for (int i = 0; i < 8; ++i) {
                v2bf pr; pr[0] = r0[i]; pr[1] = r1[i];
                *(v2bf*)(d + (n0 + i) * LDAB) = pr;
            }
        }
        __syncthreads();

        v16bf afr[2];
#pragma unroll
        for (int i = 0; i < 2; ++i)
            afr[i] = fragA(At + (wm * 32 + i * 16 + (lane & 15)) * LDAB, lane);
#pragma unroll
        for (int j = 0; j < 4; ++j) {
            v16bf bfr = fragB(Bt + (wn * 64 + j * 16 + (lane & 15)) * LDAB, lane);
#pragma unroll
            for (int i = 0; i < 2; ++i)
                acc[i][j] = __builtin_amdgcn_wmma_f32_16x16x32_bf16(
                    false, afr[i], false, bfr, (short)0, acc[i][j], false, false);
        }
    }

    float* ob = out + (size_t)b * 256 * HW;
#pragma unroll
    for (int i = 0; i < 2; ++i) {
        int ml = wm * 32 + i * 16 + (lane >> 4) * 8;
#pragma unroll
        for (int j = 0; j < 4; ++j) {
            int n = nt + wn * 64 + j * 16 + (lane & 15);
#pragma unroll
            for (int r = 0; r < 8; ++r)
                ob[(size_t)(mt + ml + r) * HW + n] = acc[i][j][r] + biasS[ml + r];
        }
    }
}

// ---------------------------------------------------------------------------
extern "C" void kernel_launch(void* const* d_in, const int* in_sizes, int n_in,
                              void* d_out, int out_size, void* d_ws, size_t ws_size,
                              hipStream_t stream) {
    (void)in_sizes; (void)n_in; (void)out_size; (void)ws_size;
    const float* x      = (const float*)d_in[0];
    const float* qkv_w  = (const float*)d_in[1];
    const float* qkv_b  = (const float*)d_in[2];
    const float* proj_w = (const float*)d_in[3];
    const float* proj_b = (const float*)d_in[4];
    float* out = (float*)d_out;

    __bf16* qkv_ws = (__bf16*)d_ws;                        // 8*768*16384 bf16 = 201.3 MB
    __bf16* y_ws   = qkv_ws + (size_t)8 * 768 * HW;        // 8*256*16384 bf16 =  67.1 MB

    const int SMEM_GEMM = 2 * 128 * LDAB * 2 + 128 * 4;    // 20992 B
    hipFuncSetAttribute(reinterpret_cast<const void*>(qkv_gemm_kernel),
                        hipFuncAttributeMaxDynamicSharedMemorySize, SMEM_GEMM);
    hipFuncSetAttribute(reinterpret_cast<const void*>(attn_kernel),
                        hipFuncAttributeMaxDynamicSharedMemorySize, SMEM_ATTN);
    hipFuncSetAttribute(reinterpret_cast<const void*>(proj_gemm_kernel),
                        hipFuncAttributeMaxDynamicSharedMemorySize, SMEM_GEMM);

    dim3 blk(256);
    qkv_gemm_kernel<<<dim3(128, 6, 8), blk, SMEM_GEMM, stream>>>(x, qkv_w, qkv_b, qkv_ws);
    attn_kernel<<<dim3(2048), blk, SMEM_ATTN, stream>>>(qkv_ws, y_ws);
    proj_gemm_kernel<<<dim3(128, 2, 8), blk, SMEM_GEMM, stream>>>(y_ws, proj_w, proj_b, out);
}
